// Generator_7679401525354
// MI455X (gfx1250) — compile-verified, matrix operand order checked by reading, not run on hardware
//
#include <hip/hip_runtime.h>

// ---------------------------------------------------------------------------
// StyleGAN2-like generator forward for MI455X (gfx1250, wave32, WMMA).
//
// conv2d_mod(x, style, W, demod) ==
//   y[b,co] = d[b,co] * conv3x3( x[b,ci] * s[b,ci],  W )      (shared W!)
//   d[b,co] = rsqrt( sum_ci Wsq[co,ci] * s[b,ci]^2 + eps )
// Heavy convs -> batch-shared-weight implicit GEMM -> v_wmma_f32_16x16x32_bf16.
// Inputs pre-scaled to zero-padded NHWC bf16; conv staging is double-buffered
// global_load_async_to_lds_b128 (ASYNCcnt) overlapped with the WMMA tap loop.
// ---------------------------------------------------------------------------

#define GEPS   1e-8f
#define IMG    128
#define GBATCH 16
#define GLAT   512

typedef __attribute__((ext_vector_type(16))) __bf16 bf16x16;
typedef __attribute__((ext_vector_type(8)))  __bf16 bf16x8;
typedef __attribute__((ext_vector_type(8)))  float  f32x8;

__device__ __forceinline__ unsigned short f2bf(float f) {
    union { float f; unsigned int u; } v; v.f = f;
    unsigned int u = v.u;
    u += 0x7FFFu + ((u >> 16) & 1u);           // round-to-nearest-even
    return (unsigned short)(u >> 16);
}

// ------------------------------ style GEMVs --------------------------------
__global__ void k_style_linear(const float* __restrict__ styles, int styleRowStride,
                               const float* __restrict__ w, const float* __restrict__ bias,
                               float* __restrict__ out, int CO, float addc) {
    int t = blockIdx.x * blockDim.x + threadIdx.x;
    if (t >= GBATCH * CO) return;
    int b = t / CO, c = t % CO;
    const float* sv = styles + (size_t)b * styleRowStride;
    const float* wv = w + (size_t)c * GLAT;
    float acc = 0.f;
#pragma unroll 8
    for (int k = 0; k < GLAT; ++k) acc += sv[k] * wv[k];
    out[t] = acc + bias[c] + addc;
}

// wsq[co,ci] = sum_tap W[co,ci,tap]^2
__global__ void k_wsq(const float* __restrict__ w, float* __restrict__ wsq, int total) {
    int t = blockIdx.x * blockDim.x + threadIdx.x;
    if (t >= total) return;
    const float* p = w + (size_t)t * 9;
    float a = 0.f;
#pragma unroll
    for (int k = 0; k < 9; ++k) a += p[k] * p[k];
    wsq[t] = a;
}

// d[b,co] = rsqrt( sum_ci wsq[co,ci] * s[b,ci]^2 + eps )
__global__ void k_demod(const float* __restrict__ s, const float* __restrict__ wsq,
                        float* __restrict__ d, int CO, int CI) {
    int t = blockIdx.x * blockDim.x + threadIdx.x;
    if (t >= GBATCH * CO) return;
    int b = t / CO, co = t % CO;
    const float* wr = wsq + (size_t)co * CI;
    const float* sr = s + (size_t)b * CI;
    float a = 0.f;
#pragma unroll 4
    for (int ci = 0; ci < CI; ++ci) { float sv = sr[ci]; a += wr[ci] * sv * sv; }
    d[t] = rsqrtf(a + GEPS);
}

// Wt[tap][co][ci] (bf16) = W[co][ci][tap] (f32)
__global__ void k_wcvt(const float* __restrict__ w, unsigned short* __restrict__ wt,
                       int CO, int CI) {
    int idx = blockIdx.x * blockDim.x + threadIdx.x;
    int total = 9 * CO * CI;
    if (idx >= total) return;
    int ci  = idx % CI;
    int rem = idx / CI;
    int co  = rem % CO;
    int tap = rem / CO;
    wt[idx] = f2bf(w[((size_t)co * CI + ci) * 9 + tap]);
}

// xs[b][py][px][c] (bf16, zero halo, NHWC padded) = x[b,c,py-1,px-1] * s[b,c]
__global__ void k_scalecvt(const float* __restrict__ x, const float* __restrict__ s,
                           unsigned short* __restrict__ xs, int C, int H) {
    int W = H;
    int total = GBATCH * (H + 2) * (W + 2) * C;
    int t = blockIdx.x * blockDim.x + threadIdx.x;
    if (t >= total) return;
    int c   = t % C;
    int rem = t / C;
    int px  = rem % (W + 2);
    rem    /= (W + 2);
    int py  = rem % (H + 2);
    int b   = rem / (H + 2);
    int gy = py - 1, gx = px - 1;
    float v = 0.f;
    if ((unsigned)gy < (unsigned)H && (unsigned)gx < (unsigned)W)
        v = x[(((size_t)b * C + c) * H + gy) * W + gx] * s[(size_t)b * C + c];
    xs[t] = f2bf(v);
}

// ---------------------- modulated 3x3 conv via WMMA ------------------------
// Per wave: D[16co x 16px] * NACC tiles. A from global Wt[tap][co][ci] (bf16),
// B patch async-copied into double-buffered LDS, pipelined with the tap loop.
template<int NACC>
__global__ void k_modconv3x3(const unsigned short* __restrict__ xs,
                             const unsigned short* __restrict__ Wt,
                             const float* __restrict__ d,
                             const float* __restrict__ nw, const float* __restrict__ nb,
                             const float* __restrict__ inoise,
                             float* __restrict__ out,
                             int CI, int CO, int H, int twShift, int th) {
    __shared__ __align__(16) unsigned short lds[2][80 * 32];  // 2 x patch(<=72) x 32ch bf16

    const int W  = H;
    const int tw = 1 << twShift;
    const int pw = tw + 2;
    const int patch = pw * (th + 2);

    const int tid  = threadIdx.x;
    const int lane = tid & 31;
    const int wave = tid >> 5;
    const int nth  = blockDim.x;
    const int wavesPer = nth >> 5;

    const int b  = blockIdx.z;
    const int tilesX = W >> twShift;
    const int y0 = (blockIdx.x / tilesX) * th;
    const int x0 = (blockIdx.x % tilesX) * tw;
    const int co0 = (blockIdx.y * wavesPer + wave) * 16;

    const int nA    = lane & 15;           // A row / D column
    const int c_off = (lane & 16) ? 8 : 0; // K-half split per ISA layout

    int pyv[NACC], pxv[NACC];
#pragma unroll
    for (int j = 0; j < NACC; ++j) {
        int n = j * 16 + nA;
        pyv[j] = n >> twShift;
        pxv[j] = n & (tw - 1);
    }

    f32x8 acc[NACC];
#pragma unroll
    for (int j = 0; j < NACC; ++j) acc[j] = (f32x8){0.f,0.f,0.f,0.f,0.f,0.f,0.f,0.f};

    const size_t rowPitch  = (size_t)(W + 2) * CI;           // elements
    const size_t basePlane = (size_t)b * (H + 2) * rowPitch;
    const int nChunks = CI >> 5;

    // issue async copies of one 32-channel K-chunk into LDS buffer p
    auto stage = [&](int ch, int p) {
        const int cbase = ch << 5;
        for (int seg = tid; seg < patch * 4; seg += nth) {
            int pp  = seg >> 2, q = seg & 3;
            int ppy = pp / pw, ppx = pp - ppy * pw;
            const unsigned short* gp = xs + basePlane
                + (size_t)(y0 + ppy) * rowPitch + (size_t)(x0 + ppx) * CI
                + cbase + q * 8;
            unsigned ldsOff = (unsigned)(unsigned long long)&lds[p][pp * 32 + q * 8];
            asm volatile("global_load_async_to_lds_b128 %0, %1, off"
                         :: "v"(ldsOff), "v"(gp) : "memory");
        }
    };

    // prologue: stage chunk 0
    stage(0, 0);
    asm volatile("s_wait_asynccnt 0x0" ::: "memory");
    __syncthreads();

    for (int ch = 0; ch < nChunks; ++ch) {
        const int p = ch & 1;
        const int cbase = ch << 5;
        // overlap: DMA next chunk into the other buffer while we do WMMAs
        if (ch + 1 < nChunks) stage(ch + 1, p ^ 1);

#pragma unroll
        for (int tap = 0; tap < 9; ++tap) {
            const int dy = tap / 3, dx = tap % 3;
            // A fragment: two contiguous 16B global loads from Wt (L2 resident)
            const unsigned short* ap =
                Wt + (((size_t)tap * CO + co0 + nA) * CI + cbase + c_off);
            bf16x8 alo = *(const bf16x8*)(const void*)ap;
            bf16x8 ahi = *(const bf16x8*)(const void*)(ap + 16);
            bf16x16 afrag = __builtin_shufflevector(alo, ahi,
                0,1,2,3,4,5,6,7,8,9,10,11,12,13,14,15);
#pragma unroll
            for (int j = 0; j < NACC; ++j) {
                int pp = (pyv[j] + dy) * pw + (pxv[j] + dx);
                const unsigned short* bp = &lds[p][pp * 32 + c_off];
                bf16x8 blo = *(const bf16x8*)(const void*)bp;
                bf16x8 bhi = *(const bf16x8*)(const void*)(bp + 16);
                bf16x16 bfrag = __builtin_shufflevector(blo, bhi,
                    0,1,2,3,4,5,6,7,8,9,10,11,12,13,14,15);
                acc[j] = __builtin_amdgcn_wmma_f32_16x16x32_bf16(
                    false, afrag, false, bfrag, (short)0, acc[j], false, false);
            }
        }

        // chunk boundary: next chunk's DMA done + all waves done reading lds[p]
        asm volatile("s_wait_asynccnt 0x0" ::: "memory");
        __syncthreads();
    }

    // ---- epilogue: demod * acc + noise, leaky-relu, store ----
#pragma unroll
    for (int j = 0; j < NACC; ++j) {
        const int gy = y0 + pyv[j];
        const int gx = x0 + pxv[j];
#pragma unroll
        for (int v = 0; v < 8; ++v) {
            int co = co0 + v + ((lane & 16) ? 8 : 0);
            float val = acc[j][v];
            val *= d[(size_t)b * CO + co];
            // noise[b,co,gy,gx] = inoise[b, gx, gy] * nw[co] + nb[co] (transposed HW)
            val += inoise[((size_t)b * IMG + gx) * IMG + gy] * nw[co] + nb[co];
            val = (val >= 0.f) ? val : 0.2f * val;
            out[(((size_t)b * CO + co) * H + gy) * W + gx] = val;
        }
    }
}

// ------------------------- 1x1 to-RGB (3 channels) -------------------------
__global__ void k_rgb(const float* __restrict__ x, const float* __restrict__ s,
                      const float* __restrict__ wr, const float* __restrict__ prev,
                      float* __restrict__ out, int CI, int HW, int hasPrev) {
    int t = blockIdx.x * blockDim.x + threadIdx.x;
    if (t >= GBATCH * 3 * HW) return;
    int pix = t % HW;
    int c3  = (t / HW) % 3;
    int b   = t / (3 * HW);
    const float* wrow = wr + (size_t)c3 * CI;
    const float* srow = s + (size_t)b * CI;
    const float* xb   = x + (size_t)b * CI * HW + pix;
    float a = 0.f;
#pragma unroll 4
    for (int ci = 0; ci < CI; ++ci) a += wrow[ci] * srow[ci] * xb[(size_t)ci * HW];
    if (hasPrev) a += prev[t];
    out[t] = a;
}

// ----------------------- bilinear 2x upsample (edge) -----------------------
__global__ void k_upsample(const float* __restrict__ in, float* __restrict__ out,
                           int planes, int H, int W) {
    int OH = 2 * H, OW = 2 * W;
    int t = blockIdx.x * blockDim.x + threadIdx.x;
    if (t >= planes * OH * OW) return;
    int c  = t % OW;
    int r  = (t / OW) % OH;
    int p  = t / (OW * OH);
    float fy = r * 0.5f - 0.25f;
    float fx = c * 0.5f - 0.25f;
    int iy = (int)floorf(fy), ix = (int)floorf(fx);
    float wy = fy - iy, wx = fx - ix;
    int y0c = iy < 0 ? 0 : (iy >= H ? H - 1 : iy);
    int y1t = iy + 1; int y1c = y1t < 0 ? 0 : (y1t >= H ? H - 1 : y1t);
    int x0c = ix < 0 ? 0 : (ix >= W ? W - 1 : ix);
    int x1t = ix + 1; int x1c = x1t < 0 ? 0 : (x1t >= W ? W - 1 : x1t);
    const float* ip = in + (size_t)p * H * W;
    float v00 = ip[y0c * W + x0c], v01 = ip[y0c * W + x1c];
    float v10 = ip[y1c * W + x0c], v11 = ip[y1c * W + x1c];
    out[t] = (1.f - wy) * ((1.f - wx) * v00 + wx * v01)
           + wy * ((1.f - wx) * v10 + wx * v11);
}

// -------------------- depthwise 3x3 blur, reflect padding ------------------
__global__ void k_blur(const float* __restrict__ in, float* __restrict__ out,
                       int planes, int H, int W) {
    int t = blockIdx.x * blockDim.x + threadIdx.x;
    if (t >= planes * H * W) return;
    int c = t % W;
    int r = (t / W) % H;
    int p = t / (W * H);
    const float* ip = in + (size_t)p * H * W;
    const float wv[3] = {0.25f, 0.5f, 0.25f};
    float a = 0.f;
#pragma unroll
    for (int ky = -1; ky <= 1; ++ky) {
        int ry = r + ky; ry = ry < 0 ? -ry : (ry >= H ? 2 * H - 2 - ry : ry);
#pragma unroll
        for (int kx = -1; kx <= 1; ++kx) {
            int rx = c + kx; rx = rx < 0 ? -rx : (rx >= W ? 2 * W - 2 - rx : rx);
            a += wv[ky + 1] * wv[kx + 1] * ip[ry * W + rx];
        }
    }
    out[t] = a;
}

// ------------------ initial 3x3 conv (batch-invariant input) ---------------
__global__ void k_init_conv(const float* __restrict__ blk, const float* __restrict__ w,
                            const float* __restrict__ bias, float* __restrict__ out) {
    int t = blockIdx.x * blockDim.x + threadIdx.x;
    if (t >= 512 * 16) return;
    int pix = t % 16, co = t / 16;
    int y = pix >> 2, x = pix & 3;
    float acc = bias[co];
    for (int ci = 0; ci < 512; ++ci) {
#pragma unroll
        for (int tap = 0; tap < 9; ++tap) {
            int dy = tap / 3 - 1, dx = tap % 3 - 1;
            int gy = y + dy, gx = x + dx;
            if ((unsigned)gy < 4u && (unsigned)gx < 4u)
                acc += blk[(ci * 4 + gy) * 4 + gx]
                     * w[((size_t)(co * 512 + ci) * 3 + (dy + 1)) * 3 + (dx + 1)];
        }
    }
    for (int b = 0; b < GBATCH; ++b)
        out[((size_t)b * 512 + co) * 16 + pix] = acc;
}

// ---------------------------------------------------------------------------
extern "C" void kernel_launch(void* const* d_in, const int* in_sizes, int n_in,
                              void* d_out, int out_size, void* d_ws, size_t ws_size,
                              hipStream_t stream) {
    (void)in_sizes; (void)n_in; (void)out_size; (void)ws_size;

    const float* styles   = (const float*)d_in[0];   // (16, 6, 512)
    const float* inoise   = (const float*)d_in[1];   // (16, 128, 128, 1)
    const float* init_blk = (const float*)d_in[2];   // (1, 512, 4, 4)
    const float* init_w   = (const float*)d_in[3];   // (512, 512, 3, 3)
    const float* init_b   = (const float*)d_in[4];   // (512,)

    // per-block params, dict keys flattened alphabetically (JAX pytree order):
    // c1_w, c2_w, n1_b, n1_w, n2_b, n2_w, rgb_ts_b, rgb_ts_w, rgb_w,
    // ts1_b, ts1_w, ts2_b, ts2_w
    struct BP { const float *c1,*c2,*n1b,*n1w,*n2b,*n2w,*rtsb,*rtsw,*rw,*ts1b,*ts1w,*ts2b,*ts2w; };
    BP bp[6];
    for (int i = 0; i < 6; ++i) {
        const float** q = (const float**)&bp[i];
        for (int k = 0; k < 13; ++k) q[k] = (const float*)d_in[5 + 13 * i + k];
    }

    static const int CIs[6] = {512, 512, 512, 256, 128, 64};
    static const int COs[6] = {512, 512, 256, 128, 64, 32};

    // ---- workspace carve ----
    char* ws = (char*)d_ws;
    size_t off = 0;
    auto carve = [&](size_t bytes) -> void* {
        void* p = ws + off;
        off += (bytes + 255) & ~(size_t)255;
        return p;
    };
    float* xA   = (float*)carve(64ull << 20);                    // up to 16x64x128x128 f32
    float* xB   = (float*)carve(64ull << 20);
    float* rgbA = (float*)carve((size_t)GBATCH * 3 * IMG * IMG * 4);
    float* rgbB = (float*)carve((size_t)GBATCH * 3 * IMG * IMG * 4);
    float* s1   = (float*)carve((size_t)GBATCH * 512 * 4);
    float* s2   = (float*)carve((size_t)GBATCH * 512 * 4);
    float* sr   = (float*)carve((size_t)GBATCH * 512 * 4);
    float* dd1  = (float*)carve((size_t)GBATCH * 512 * 4);
    float* dd2  = (float*)carve((size_t)GBATCH * 512 * 4);
    float* wsq  = (float*)carve((size_t)512 * 512 * 4);
    unsigned short* Wt = (unsigned short*)carve((size_t)9 * 512 * 512 * 2);
    unsigned short* xs = (unsigned short*)carve(40ull << 20);    // NHWC padded bf16

    const int TB = 256;
    auto nb = [](int n, int t) { return (n + t - 1) / t; };

    // ---- initial block + init conv ----
    k_init_conv<<<nb(512 * 16, TB), TB, 0, stream>>>(init_blk, init_w, init_b, xA);

    float* cur = xA;
    float* oth = xB;
    float* rgbPrev = nullptr;
    int H = 4;

    for (int i = 0; i < 6; ++i) {
        const int CI = CIs[i], CO = COs[i];
        const BP& p = bp[i];
        const float* styRow = styles + (size_t)i * GLAT;       // batch stride 6*512

        if (i > 0) {
            k_upsample<<<nb(GBATCH * CI * 4 * H * H, TB), TB, 0, stream>>>(
                cur, oth, GBATCH * CI, H, H);
            float* t = cur; cur = oth; oth = t;
            H *= 2;
        }

        // tile geometry: H>=16 -> 16x2 tiles (32px, 2 acc); H==8 -> 8x4; H==4 -> 4x4 (16px)
        const int twShift = (H >= 16) ? 4 : (H == 8 ? 3 : 2);
        const int th      = (H >= 16) ? 2 : 4;
        const int nacc    = (H == 4) ? 1 : 2;
        const int wavesPer = (CO >= 64) ? 4 : (CO / 16);
        dim3 cgrid(H * H / (16 * nacc), CO / (16 * wavesPer), GBATCH);
        dim3 cblk(32 * wavesPer);
        const int padTotC1 = GBATCH * (H + 2) * (H + 2) * CI;
        const int padTotC2 = GBATCH * (H + 2) * (H + 2) * CO;

        // conv1 (ci -> co): s1 / d1 / noise1
        k_style_linear<<<nb(GBATCH * CI, TB), TB, 0, stream>>>(
            styRow, 6 * GLAT, p.ts1w, p.ts1b, s1, CI, 1.0f);
        k_wsq<<<nb(CO * CI, TB), TB, 0, stream>>>(p.c1, wsq, CO * CI);
        k_demod<<<nb(GBATCH * CO, TB), TB, 0, stream>>>(s1, wsq, dd1, CO, CI);
        k_wcvt<<<nb(9 * CO * CI, TB), TB, 0, stream>>>(p.c1, Wt, CO, CI);
        k_scalecvt<<<nb(padTotC1, TB), TB, 0, stream>>>(cur, s1, xs, CI, H);
        if (nacc == 1)
            k_modconv3x3<1><<<cgrid, cblk, 0, stream>>>(
                xs, Wt, dd1, p.n1w, p.n1b, inoise, oth, CI, CO, H, twShift, th);
        else
            k_modconv3x3<2><<<cgrid, cblk, 0, stream>>>(
                xs, Wt, dd1, p.n1w, p.n1b, inoise, oth, CI, CO, H, twShift, th);

        // conv2 (co -> co): s2 / d2 / noise2
        k_style_linear<<<nb(GBATCH * CO, TB), TB, 0, stream>>>(
            styRow, 6 * GLAT, p.ts2w, p.ts2b, s2, CO, 1.0f);
        k_wsq<<<nb(CO * CO, TB), TB, 0, stream>>>(p.c2, wsq, CO * CO);
        k_demod<<<nb(GBATCH * CO, TB), TB, 0, stream>>>(s2, wsq, dd2, CO, CO);
        k_wcvt<<<nb(9 * CO * CO, TB), TB, 0, stream>>>(p.c2, Wt, CO, CO);
        k_scalecvt<<<nb(padTotC2, TB), TB, 0, stream>>>(oth, s2, xs, CO, H);
        if (nacc == 1)
            k_modconv3x3<1><<<cgrid, cblk, 0, stream>>>(
                xs, Wt, dd2, p.n2w, p.n2b, inoise, cur, CO, CO, H, twShift, th);
        else
            k_modconv3x3<2><<<cgrid, cblk, 0, stream>>>(
                xs, Wt, dd2, p.n2w, p.n2b, inoise, cur, CO, CO, H, twShift, th);
        // x now lives in `cur` with CO channels (== CI of next block)

        // to-RGB (1x1, modulated, no demod) + residual add
        k_style_linear<<<nb(GBATCH * CO, TB), TB, 0, stream>>>(
            styRow, 6 * GLAT, p.rtsw, p.rtsb, sr, CO, 1.0f);
        float* rgbOut = (i == 5) ? (float*)d_out : rgbA;
        k_rgb<<<nb(GBATCH * 3 * H * H, TB), TB, 0, stream>>>(
            cur, sr, p.rw, rgbPrev ? rgbPrev : rgbA, rgbOut, CO, H * H, rgbPrev ? 1 : 0);

        if (i < 5) {
            // rgb = blur(upsample2x(rgb))
            k_upsample<<<nb(GBATCH * 3 * 4 * H * H, TB), TB, 0, stream>>>(
                rgbA, rgbB, GBATCH * 3, H, H);
            k_blur<<<nb(GBATCH * 3 * 4 * H * H, TB), TB, 0, stream>>>(
                rgbB, rgbA, GBATCH * 3, 2 * H, 2 * H);
            rgbPrev = rgbA;
        }
    }
}
